// WindowAttention_23605140259509
// MI455X (gfx1250) — matrix operand order repaired and checked by reading, not compiled, not run on hardware
//
#include <hip/hip_runtime.h>
#include <hip/hip_bf16.h>

typedef _Float16 h16;
typedef __attribute__((ext_vector_type(8)))  _Float16 v8h;
typedef __attribute__((ext_vector_type(16))) _Float16 v16h;
typedef __attribute__((ext_vector_type(8)))  float    v8f;

#define N_TOK 49
#define C_DIM 128
#define NHEAD 4
#define NW    64
#define BWIN  4096

// workspace layout (bytes)
#define WS_WTQ  0                        // 384*128 f16 (qkv_w transposed)
#define WS_WTP  (384*128*2)              // 128*128 f16 (proj_w transposed)
#define WS_BIAS (WS_WTP + 128*128*2)     // 4*49*49 f32 (rel-pos bias per head)

// LDS layout (in halves)
#define XA_OFF  0          // [64][136]  x as f16 (also reused as ctx in phase 3/4)
#define Q_OFF   8704       // [4][64][40]
#define K_OFF   18944      // [4][64][40]
#define VT_OFF  29184      // [4][32][72] (v transposed)
#define AP_OFF  38400      // [4][64][72] (attn probs)
#define LDS_HALVES 56832   // 113664 bytes

// ---- fragment loaders (layouts per CDNA5 ISA 7.12.2) ----
// A (16-bit 16x32): lanes 0-15 row M=lane, K chunks {0..7,16..23}; lanes 16-31 K {8..15,24..31}
__device__ __forceinline__ v16h ldfragA(const h16* base, int stride, int row0, int k0, int lane) {
  int r = lane & 15, hi = lane >> 4;
  const h16* p = base + (row0 + r) * stride + k0 + hi * 8;
  v8h c0 = *(const v8h*)p;
  v8h c1 = *(const v8h*)(p + 16);
  return __builtin_shufflevector(c0, c1, 0,1,2,3,4,5,6,7,8,9,10,11,12,13,14,15);
}
// B (16-bit 32x16): lane holds column N=lane%16; lanes 0-15 K {0..7,16..23}, lanes 16-31 K {8..15,24..31}.
// 'base' is B^T row-major (row = output column), so the 16 K-values are contiguous.
__device__ __forceinline__ v16h ldfragB(const h16* base, int stride, int n0, int k0, int lane) {
  int n = n0 + (lane & 15), hi = lane >> 4;
  const h16* p = base + n * stride + k0 + hi * 8;
  v8h c0 = *(const v8h*)p;
  v8h c1 = *(const v8h*)(p + 16);
  return __builtin_shufflevector(c0, c1, 0,1,2,3,4,5,6,7,8,9,10,11,12,13,14,15);
}
__device__ __forceinline__ v8f wmma_f16(v16h a, v16h b, v8f c) {
  return __builtin_amdgcn_wmma_f32_16x16x32_f16(false, a, false, b, (short)0, c, false, false);
}

// ---- one-time prep: transpose+convert weights to f16, gather rel-pos bias ----
__global__ __launch_bounds__(256)
void prep_kernel(const float* __restrict__ qkv_w, const float* __restrict__ proj_w,
                 const float* __restrict__ bias_table, const int* __restrict__ rel_index,
                 h16* __restrict__ wtq, h16* __restrict__ wtp, float* __restrict__ biasf) {
  int idx = blockIdx.x * 256 + threadIdx.x;
  if (idx < 384 * 128) {               // wtq[n][k] = qkv_w[k][n]
    int n = idx >> 7, k = idx & 127;
    wtq[idx] = (h16)qkv_w[k * 384 + n];
  }
  if (idx < 128 * 128) {               // wtp[n][k] = proj_w[k][n]
    int n = idx >> 7, k = idx & 127;
    wtp[idx] = (h16)proj_w[k * 128 + n];
  }
  if (idx < NHEAD * N_TOK * N_TOK) {   // biasf[h][n][m] = bias_table[rel_index[n][m]][h]
    int h = idx / (N_TOK * N_TOK), rem = idx % (N_TOK * N_TOK);
    biasf[idx] = bias_table[rel_index[rem] * NHEAD + h];
  }
}

// ---- main kernel: one workgroup (4 wave32) per window ----
__global__ __launch_bounds__(128)
void win_attn_kernel(const float* __restrict__ x, const float* __restrict__ amask,
                     const float* __restrict__ qkv_b, const float* __restrict__ proj_b,
                     const h16* __restrict__ wtq, const h16* __restrict__ wtp,
                     const float* __restrict__ biasf, float* __restrict__ out) {
  extern __shared__ h16 smem[];
  h16* xA  = smem + XA_OFF;   // [64][136]
  h16* qS  = smem + Q_OFF;    // [4][64][40]
  h16* kS  = smem + K_OFF;    // [4][64][40]
  h16* vT  = smem + VT_OFF;   // [4][32][72]
  h16* aP  = smem + AP_OFF;   // [4][64][72]
  h16* ctx = smem + XA_OFF;   // reuse xA region: [64][136]

  const int b    = blockIdx.x;
  const int w    = b & (NW - 1);
  const int tid  = threadIdx.x;
  const int wave = tid >> 5;
  const int lane = tid & 31;
  const int hi   = lane >> 4;
  const int cl   = lane & 15;

  // Phase 0: x[b] (f32) -> xA f16, rows 49..63 zeroed
  {
    const float4* xv = (const float4*)(x + (size_t)b * N_TOK * C_DIM);
    for (int i = tid; i < 64 * 32; i += 128) {
      int row = i >> 5, c4 = i & 31;
      float4 v = (row < N_TOK) ? xv[row * 32 + c4] : make_float4(0.f, 0.f, 0.f, 0.f);
      h16* d = xA + row * 136 + c4 * 4;
      d[0] = (h16)v.x; d[1] = (h16)v.y; d[2] = (h16)v.z; d[3] = (h16)v.w;
    }
  }
  __syncthreads();

  // Phase 1: qkv = x @ qkv_w + qkv_b ; wave handles row-tile mt=wave, 24 col tiles
  {
    const int mt = wave;
    v16h afr[4];
    #pragma unroll
    for (int kc = 0; kc < 4; ++kc) afr[kc] = ldfragA(xA, 136, mt * 16, kc * 32, lane);
    for (int ct = 0; ct < 24; ++ct) {
      v8f acc = {};
      #pragma unroll
      for (int kc = 0; kc < 4; ++kc)
        acc = wmma_f16(afr[kc], ldfragB(wtq, 128, ct * 16, kc * 32, lane), acc);
      int col  = ct * 16 + cl;           // 0..383
      float bv = qkv_b[col];
      int jt   = col >> 7;               // 0=q 1=k 2=v
      int c128 = col & 127;
      int h    = c128 >> 5;
      int d    = c128 & 31;
      #pragma unroll
      for (int r = 0; r < 8; ++r) {
        int row = mt * 16 + r + hi * 8;
        h16 val = (h16)(acc[r] + bv);
        if (jt == 0)      qS[h * 2560 + row * 40 + d] = val;
        else if (jt == 1) kS[h * 2560 + row * 40 + d] = val;
        else              vT[h * 2304 + d * 72 + row] = val;   // transposed
      }
    }
  }
  __syncthreads();

  // Phase 2: logits = scale*q@k^T + bias + mask ; softmax ; wave = head
  {
    const int h = wave;
    const float scale = 0.17677669529663687f;   // 32^-0.5
    const float* biash = biasf + h * (N_TOK * N_TOK);
    const float* maskw = amask + w * (N_TOK * N_TOK);
    for (int mt = 0; mt < 4; ++mt) {
      v16h a = ldfragA(qS + h * 2560, 40, mt * 16, 0, lane);
      v8f acc[4];
      #pragma unroll
      for (int nt = 0; nt < 4; ++nt) {
        v8f z = {};
        acc[nt] = wmma_f16(a, ldfragB(kS + h * 2560, 40, nt * 16, 0, lane), z);
      }
      #pragma unroll
      for (int r = 0; r < 8; ++r) {
        int row = mt * 16 + r + hi * 8;
        float vals[4];
        #pragma unroll
        for (int nt = 0; nt < 4; ++nt) {
          int col = nt * 16 + cl;
          vals[nt] = (row < N_TOK && col < N_TOK)
                   ? acc[nt][r] * scale + biash[row * N_TOK + col] + maskw[row * N_TOK + col]
                   : -1e30f;
        }
        float m = fmaxf(fmaxf(vals[0], vals[1]), fmaxf(vals[2], vals[3]));
        for (int s = 1; s < 16; s <<= 1) m = fmaxf(m, __shfl_xor(m, s, 32));
        float e[4], sum = 0.f;
        #pragma unroll
        for (int nt = 0; nt < 4; ++nt) { e[nt] = __expf(vals[nt] - m); sum += e[nt]; }
        for (int s = 1; s < 16; s <<= 1) sum += __shfl_xor(sum, s, 32);
        float inv = 1.0f / sum;
        #pragma unroll
        for (int nt = 0; nt < 4; ++nt)
          aP[h * 4608 + row * 72 + nt * 16 + cl] = (h16)(e[nt] * inv);
      }
    }
  }
  __syncthreads();

  // Phase 3: ctx = attnP @ v ; wave = head
  {
    const int h = wave;
    v16h afr[4][2];
    #pragma unroll
    for (int mt = 0; mt < 4; ++mt)
      #pragma unroll
      for (int kc = 0; kc < 2; ++kc)
        afr[mt][kc] = ldfragA(aP + h * 4608, 72, mt * 16, kc * 32, lane);
    #pragma unroll
    for (int dt = 0; dt < 2; ++dt) {
      v16h b0 = ldfragB(vT + h * 2304, 72, dt * 16, 0,  lane);
      v16h b1 = ldfragB(vT + h * 2304, 72, dt * 16, 32, lane);
      #pragma unroll
      for (int mt = 0; mt < 4; ++mt) {
        v8f acc = {};
        acc = wmma_f16(afr[mt][0], b0, acc);
        acc = wmma_f16(afr[mt][1], b1, acc);
        #pragma unroll
        for (int r = 0; r < 8; ++r) {
          int row = mt * 16 + r + hi * 8;
          ctx[row * 136 + h * 32 + dt * 16 + cl] = (h16)acc[r];
        }
      }
    }
  }
  __syncthreads();

  // Phase 4: out = ctx @ proj_w + proj_b ; wave handles row-tile mt=wave
  {
    const int mt = wave;
    v16h afr[4];
    #pragma unroll
    for (int kc = 0; kc < 4; ++kc) afr[kc] = ldfragA(ctx, 136, mt * 16, kc * 32, lane);
    float* ob = out + (size_t)b * N_TOK * C_DIM;
    for (int ct = 0; ct < 8; ++ct) {
      v8f acc = {};
      #pragma unroll
      for (int kc = 0; kc < 4; ++kc)
        acc = wmma_f16(afr[kc], ldfragB(wtp, 128, ct * 16, kc * 32, lane), acc);
      int col  = ct * 16 + cl;
      float pb = proj_b[col];
      #pragma unroll
      for (int r = 0; r < 8; ++r) {
        int row = mt * 16 + r + hi * 8;
        if (row < N_TOK) ob[row * 128 + col] = acc[r] + pb;
      }
    }
  }
}

extern "C" void kernel_launch(void* const* d_in, const int* in_sizes, int n_in,
                              void* d_out, int out_size, void* d_ws, size_t ws_size,
                              hipStream_t stream) {
  const float* x      = (const float*)d_in[0];
  const float* amask  = (const float*)d_in[1];
  const float* qkv_w  = (const float*)d_in[2];
  const float* qkv_b  = (const float*)d_in[3];
  const float* btab   = (const float*)d_in[4];
  const float* proj_w = (const float*)d_in[5];
  const float* proj_b = (const float*)d_in[6];
  const int*   relidx = (const int*)d_in[7];
  float* out = (float*)d_out;

  char* ws = (char*)d_ws;
  h16*   wtq   = (h16*)(ws + WS_WTQ);
  h16*   wtp   = (h16*)(ws + WS_WTP);
  float* biasf = (float*)(ws + WS_BIAS);

  prep_kernel<<<192, 256, 0, stream>>>(qkv_w, proj_w, btab, relidx, wtq, wtp, biasf);

  const size_t smem = (size_t)LDS_HALVES * sizeof(h16);   // 113664 B (WGP LDS: 320 KB)
  (void)hipFuncSetAttribute((const void*)win_attn_kernel,
                            hipFuncAttributeMaxDynamicSharedMemorySize, (int)smem);
  win_attn_kernel<<<BWIN, 128, smem, stream>>>(x, amask, qkv_b, proj_b, wtq, wtp, biasf, out);
}
